// SimpleMoEModel_52166672777636
// MI455X (gfx1250) — matrix operand order repaired and checked by reading, not compiled, not run on hardware
//
#include <hip/hip_runtime.h>

// ---------------------------------------------------------------------------
// MoE block for MI455X (gfx1250, wave32, WMMA).
// Pipeline:
//   1. convert x -> bf16; transpose+convert W_in, W1[e], W2[e], W_out -> bf16
//      (weights stored [N,K] so WMMA B fragments are contiguous 16B loads)
//   2. GEMM0: h = x @ W_in + b_in            (bf16 out)
//   3. gating: softmax/top-2 per token, per-expert counts
//   4. scan + assign: compact token slots per expert (total slots = N*2)
//   5. GEMM1: mid[slot] = relu(h[tok] @ W1[e] + b1[e])   (gathered A, bf16 out)
//   6. GEMM2: moe[tok] += w_slot * (mid[slot] @ W2[e] + b2[e])  (f32 atomics)
//   7. GEMM3: out = moe @ W_out + b_out      (f32 out)
// Core GEMM: 128x128x32 tile, 8 waves, v_wmma_f32_16x16x32_bf16.
// K-tile staging uses CDNA5 async global->LDS DMA (ASYNCcnt) with LDS
// double-buffering when the builtins are available; falls back to the
// synchronous VGPR-staged path otherwise.
// ---------------------------------------------------------------------------

typedef __attribute__((ext_vector_type(16))) __bf16 v16bf;
typedef __attribute__((ext_vector_type(8)))  float  v8f;

#define DIN   1024
#define DH    1024
#define DOUT  1024
#define DMOE  4096
#define NEXP  8
#define NTOK  4096
#define TOTAL_SLOTS (NTOK * 2)

#if defined(__HIP_DEVICE_COMPILE__) && defined(__has_builtin)
#if __has_builtin(__builtin_amdgcn_global_load_async_to_lds_b128) && \
    __has_builtin(__builtin_amdgcn_s_wait_asynccnt)
#define USE_ASYNC_LDS 1
#endif
#endif
#ifndef USE_ASYNC_LDS
#define USE_ASYNC_LDS 0
#endif

#if USE_ASYNC_LDS
typedef int v4i_vs __attribute__((vector_size(16)));
typedef __attribute__((address_space(1))) v4i_vs glb_v4i;
typedef __attribute__((address_space(3))) v4i_vs lds_v4i;
#define ASYNC_COPY16(dst, src)                                                 \
  __builtin_amdgcn_global_load_async_to_lds_b128((glb_v4i*)(src),              \
                                                 (lds_v4i*)(dst), 0, 0)
#endif

static __device__ __forceinline__ unsigned short f2bf(float f) {
  unsigned int u = __float_as_uint(f);
  u += 0x7FFFu + ((u >> 16) & 1u);           // round-to-nearest-even
  return (unsigned short)(u >> 16);
}
static __device__ __forceinline__ float bf2f(unsigned short s) {
  return __uint_as_float(((unsigned int)s) << 16);
}

// ------------------------------ converters ---------------------------------

__global__ __launch_bounds__(256) void cvt_f32_bf16(
    const float* __restrict__ src, unsigned short* __restrict__ dst, long long n) {
  long long i = ((long long)blockIdx.x * blockDim.x + threadIdx.x) * 4;
  if (i >= n) return;
  float4 v = *(const float4*)(src + i);
  union { unsigned short s[4]; uint2 u; } o;
  o.s[0] = f2bf(v.x); o.s[1] = f2bf(v.y); o.s[2] = f2bf(v.z); o.s[3] = f2bf(v.w);
  *(uint2*)(dst + i) = o.u;
}

// src: [z][R][C] f32  ->  dst: [z][C][R] bf16
__global__ __launch_bounds__(256) void transpose_f32_bf16(
    const float* __restrict__ src, unsigned short* __restrict__ dst, int R, int C) {
  __shared__ float tile[32][33];
  long long boff = (long long)blockIdx.z * R * C;
  int c0 = blockIdx.x * 32, r0 = blockIdx.y * 32;
  int tx = threadIdx.x & 31, ty = threadIdx.x >> 5;   // 32x8
#pragma unroll
  for (int i = 0; i < 4; ++i) {
    int r = r0 + ty + i * 8;
    tile[ty + i * 8][tx] = src[boff + (long long)r * C + c0 + tx];
  }
  __syncthreads();
#pragma unroll
  for (int i = 0; i < 4; ++i) {
    int c = c0 + ty + i * 8;
    dst[boff + (long long)c * R + r0 + tx] = f2bf(tile[tx][ty + i * 8]);
  }
}

// ------------------------------ gating -------------------------------------

__global__ __launch_bounds__(256) void gate_kernel(
    const unsigned short* __restrict__ h,      // [NTOK, DH] bf16
    const float* __restrict__ Wg,              // [DH, NEXP]
    const float* __restrict__ bg,              // [NEXP]
    int* __restrict__ topi, float* __restrict__ topw,
    int* __restrict__ counts) {
  int tok  = (int)((blockIdx.x * 256 + threadIdx.x) >> 5);   // one wave per token
  int lane = threadIdx.x & 31;
  if (tok >= NTOK) return;
  const unsigned short* hr = h + (long long)tok * DH;
  float acc[NEXP];
#pragma unroll
  for (int e = 0; e < NEXP; ++e) acc[e] = 0.f;
  for (int k = lane; k < DH; k += 32) {
    float hv = bf2f(hr[k]);
    const float* wr = Wg + k * NEXP;
#pragma unroll
    for (int e = 0; e < NEXP; ++e) acc[e] += hv * wr[e];
  }
#pragma unroll
  for (int off = 16; off >= 1; off >>= 1) {
#pragma unroll
    for (int e = 0; e < NEXP; ++e) acc[e] += __shfl_xor(acc[e], off, 32);
  }
  if (lane == 0) {
    float mx = -3.4e38f;
#pragma unroll
    for (int e = 0; e < NEXP; ++e) { acc[e] += bg[e]; mx = fmaxf(mx, acc[e]); }
    float p[NEXP], s = 0.f;
#pragma unroll
    for (int e = 0; e < NEXP; ++e) { p[e] = __expf(acc[e] - mx); s += p[e]; }
    float inv = 1.f / s;
#pragma unroll
    for (int e = 0; e < NEXP; ++e) p[e] *= inv;
    int i0 = 0;
#pragma unroll
    for (int e = 1; e < NEXP; ++e) if (p[e] > p[i0]) i0 = e;
    float v0 = p[i0]; p[i0] = -1.f;
    int i1 = 0;
#pragma unroll
    for (int e = 1; e < NEXP; ++e) if (p[e] > p[i1]) i1 = e;
    float v1 = p[i1];
    float rs = 1.f / (v0 + v1);
    topi[2 * tok] = i0; topi[2 * tok + 1] = i1;
    topw[2 * tok] = v0 * rs; topw[2 * tok + 1] = v1 * rs;
    atomicAdd(&counts[i0], 1);
    atomicAdd(&counts[i1], 1);
  }
}

__global__ void scan_kernel(const int* __restrict__ counts,
                            int* __restrict__ offsets, int* __restrict__ counters2) {
  if (threadIdx.x == 0 && blockIdx.x == 0) {
    int off = 0;
    for (int e = 0; e < NEXP; ++e) { offsets[e] = off; off += counts[e]; counters2[e] = 0; }
  }
}

__global__ __launch_bounds__(256) void assign_kernel(
    const int* __restrict__ topi, const float* __restrict__ topw,
    const int* __restrict__ offsets, int* __restrict__ counters2,
    int* __restrict__ tokOf, float* __restrict__ wOf) {
  int n = blockIdx.x * blockDim.x + threadIdx.x;
  if (n >= NTOK) return;
#pragma unroll
  for (int k = 0; k < 2; ++k) {
    int e = topi[2 * n + k];
    int p = atomicAdd(&counters2[e], 1);
    int s = offsets[e] + p;
    tokOf[s] = n;
    wOf[s] = topw[2 * n + k];
  }
}

// ------------------------------ WMMA GEMM ----------------------------------
// C[M,N] = A[M,K](bf16) x B[N,K](bf16, pre-transposed) + bias
// MODE 0: store bf16            (h = x W_in + b_in)
// MODE 1: gather A rows via token list, ReLU, store bf16 at slot rows (FFN1)
// MODE 2: A rows are slots; atomicAdd w*(acc+bias) into moe[tok] (FFN2)
// MODE 3: store f32             (out = moe W_out + b_out)
template <int MODE>
__global__ __launch_bounds__(256) void gemm_bf16(
    const unsigned short* __restrict__ A,
    const unsigned short* __restrict__ B,
    const float* __restrict__ bias,
    void* __restrict__ OUT,
    int M, int N, int K,
    long long Bstride, long long biasStride,
    const int* __restrict__ counts,
    const int* __restrict__ offsets,
    const int* __restrict__ tokOf,
    const float* __restrict__ wOf,
    float* __restrict__ moeOut) {
#if USE_ASYNC_LDS
  __shared__ __align__(16) unsigned short lA[2][128 * 40];  // pad 32->40 elems
  __shared__ __align__(16) unsigned short lB[2][128 * 40];
#else
  __shared__ __align__(16) unsigned short lA[1][128 * 40];
  __shared__ __align__(16) unsigned short lB[1][128 * 40];
#endif

  const int e  = blockIdx.z;
  const int m0 = blockIdx.x * 128;
  const int n0 = blockIdx.y * 128;

  int mcount = M, slotbase = 0;
  if (MODE == 1 || MODE == 2) {
    mcount = counts[e];
    slotbase = offsets[e];
    if (m0 >= mcount) return;
  }
  const unsigned short* Be = B + (long long)e * Bstride;
  const float* be = bias + (long long)e * biasStride;

  // -------- precompute global->LDS staging (rows fixed across K-loop) -----
  const unsigned short* aSrc[2];
  const unsigned short* bSrc[2];
  int sDst[2];
#pragma unroll
  for (int i = 0; i < 2; ++i) {
    int c  = threadIdx.x + i * 256;
    int r  = c >> 2;                 // 128 rows, 4 chunks/row
    int cc = (c & 3) * 8;            // bf16 column of 16B chunk
    long long arow;
    if (MODE == 1) {
      int rr = m0 + r; if (rr > mcount - 1) rr = mcount - 1;
      arow = tokOf[slotbase + rr];
    } else if (MODE == 2) {
      long long s = slotbase + m0 + r;
      if (s > TOTAL_SLOTS - 1) s = TOTAL_SLOTS - 1;
      arow = s;
    } else {
      arow = m0 + r;
    }
    aSrc[i] = A + arow * K + cc;
    bSrc[i] = Be + (long long)(n0 + r) * K + cc;
    sDst[i] = r * 40 + cc;
  }

  const int lane = threadIdx.x & 31;
  const int wid  = threadIdx.x >> 5;
  const int wm   = wid >> 2;          // 0..1  -> 64-row patch
  const int wn   = wid & 3;           // 0..3  -> 32-col patch
  const int half = lane >> 4;
  const int l15  = lane & 15;

  v8f acc[4][2];
#pragma unroll
  for (int f = 0; f < 4; ++f)
#pragma unroll
    for (int g = 0; g < 2; ++g)
      acc[f][g] = v8f{0.f, 0.f, 0.f, 0.f, 0.f, 0.f, 0.f, 0.f};

  const int aSeg = half * 8;          // A lane: K segs {s..s+7, s+16..s+23}
  const int bSeg = half * 8;          // B lane: K segs {s..s+7, s+16..s+23}

  const int nIter = K >> 5;

#if USE_ASYNC_LDS
  // prefetch tile 0 into buffer 0 via async DMA (no VGPR staging)
#pragma unroll
  for (int i = 0; i < 2; ++i) {
    ASYNC_COPY16(&lA[0][sDst[i]], aSrc[i]);
    ASYNC_COPY16(&lB[0][sDst[i]], bSrc[i]);
  }
  __builtin_amdgcn_s_wait_asynccnt(0);
  __syncthreads();
#endif

  for (int it = 0; it < nIter; ++it) {
    const int k0 = it << 5;
#if USE_ASYNC_LDS
    const int cur = it & 1;
    if (it + 1 < nIter) {              // prefetch next tile into other buffer
#pragma unroll
      for (int i = 0; i < 2; ++i) {
        ASYNC_COPY16(&lA[cur ^ 1][sDst[i]], aSrc[i] + k0 + 32);
        ASYNC_COPY16(&lB[cur ^ 1][sDst[i]], bSrc[i] + k0 + 32);
      }
    }
    const unsigned short* pA = lA[cur];
    const unsigned short* pB = lB[cur];
#else
#pragma unroll
    for (int i = 0; i < 2; ++i) {
      *(uint4*)&lA[0][sDst[i]] = *(const uint4*)(aSrc[i] + k0);
      *(uint4*)&lB[0][sDst[i]] = *(const uint4*)(bSrc[i] + k0);
    }
    __syncthreads();
    const unsigned short* pA = lA[0];
    const unsigned short* pB = lB[0];
#endif

    union FragU { v16bf v; uint4 q[2]; } fa[4], fb[2];
#pragma unroll
    for (int f = 0; f < 4; ++f) {
      const unsigned short* p = &pA[(wm * 64 + f * 16 + l15) * 40 + aSeg];
      fa[f].q[0] = *(const uint4*)p;
      fa[f].q[1] = *(const uint4*)(p + 16);
    }
#pragma unroll
    for (int g = 0; g < 2; ++g) {
      const unsigned short* p = &pB[(wn * 32 + g * 16 + l15) * 40 + bSeg];
      fb[g].q[0] = *(const uint4*)p;
      fb[g].q[1] = *(const uint4*)(p + 16);
    }
#pragma unroll
    for (int f = 0; f < 4; ++f)
#pragma unroll
      for (int g = 0; g < 2; ++g)
        acc[f][g] = __builtin_amdgcn_wmma_f32_16x16x32_bf16(
            false, fa[f].v, false, fb[g].v, (short)0, acc[f][g], false, false);

#if USE_ASYNC_LDS
    __builtin_amdgcn_s_wait_asynccnt(0);   // next-tile DMA landed
    __syncthreads();                        // all waves done reading cur
#else
    __syncthreads();
#endif
  }

  // ------------------------------ epilogue --------------------------------
#pragma unroll
  for (int f = 0; f < 4; ++f) {
#pragma unroll
    for (int v = 0; v < 8; ++v) {
      int mrow = m0 + wm * 64 + f * 16 + v + 8 * half;
      if ((MODE == 1 || MODE == 2) && mrow >= mcount) continue;
      int tok = 0; float w = 0.f;
      if (MODE == 2) {
        int s = slotbase + mrow;
        tok = tokOf[s];
        w = wOf[s];
      }
#pragma unroll
      for (int g = 0; g < 2; ++g) {
        int col = n0 + wn * 32 + g * 16 + l15;
        float val = acc[f][g][v] + be[col];
        if (MODE == 0) {
          ((unsigned short*)OUT)[(long long)mrow * N + col] = f2bf(val);
        } else if (MODE == 1) {
          if (val < 0.f) val = 0.f;
          ((unsigned short*)OUT)[(long long)(slotbase + mrow) * N + col] = f2bf(val);
        } else if (MODE == 2) {
          atomicAdd(&moeOut[(long long)tok * N + col], w * val);
        } else {
          ((float*)OUT)[(long long)mrow * N + col] = val;
        }
      }
    }
  }
}

// ------------------------------ launcher -----------------------------------

extern "C" void kernel_launch(void* const* d_in, const int* in_sizes, int n_in,
                              void* d_out, int out_size, void* d_ws, size_t ws_size,
                              hipStream_t stream) {
  (void)in_sizes; (void)n_in; (void)out_size; (void)ws_size;

  const float* x     = (const float*)d_in[0];
  const float* W_in  = (const float*)d_in[1];
  const float* b_in  = (const float*)d_in[2];
  const float* W_g   = (const float*)d_in[3];
  const float* b_g   = (const float*)d_in[4];
  const float* W1    = (const float*)d_in[5];
  const float* b1    = (const float*)d_in[6];
  const float* W2    = (const float*)d_in[7];
  const float* b2    = (const float*)d_in[8];
  const float* W_out = (const float*)d_in[9];
  const float* b_out = (const float*)d_in[10];

  char* ws = (char*)d_ws;
  size_t off = 0;
  auto alloc = [&](size_t bytes) -> void* {
    void* p = ws + off;
    off += (bytes + 255) & ~(size_t)255;
    return p;
  };

  unsigned short* x_bf   = (unsigned short*)alloc((size_t)NTOK * DIN * 2);
  unsigned short* WinT   = (unsigned short*)alloc((size_t)DH * DIN * 2);
  unsigned short* W1T    = (unsigned short*)alloc((size_t)NEXP * DMOE * DH * 2);
  unsigned short* W2T    = (unsigned short*)alloc((size_t)NEXP * DH * DMOE * 2);
  unsigned short* WoutT  = (unsigned short*)alloc((size_t)DOUT * DH * 2);
  unsigned short* h_bf   = (unsigned short*)alloc((size_t)NTOK * DH * 2);
  unsigned short* mid_bf = (unsigned short*)alloc((size_t)TOTAL_SLOTS * DMOE * 2);
  float*          moe    = (float*)alloc((size_t)NTOK * DH * 4);
  unsigned short* moe_bf = (unsigned short*)alloc((size_t)NTOK * DH * 2);
  int*   topi     = (int*)alloc((size_t)NTOK * 2 * 4);
  float* topw     = (float*)alloc((size_t)NTOK * 2 * 4);
  int*   counts   = (int*)alloc(NEXP * 4);
  int*   offsets  = (int*)alloc(NEXP * 4);
  int*   counters = (int*)alloc(NEXP * 4);
  int*   tokOf    = (int*)alloc((size_t)TOTAL_SLOTS * 4);
  float* wOf      = (float*)alloc((size_t)TOTAL_SLOTS * 4);

  (void)hipMemsetAsync(counts, 0, NEXP * 4, stream);
  (void)hipMemsetAsync(moe, 0, (size_t)NTOK * DH * 4, stream);

  // --- bf16 conversions (x elementwise; weights transposed to [N,K]) ------
  {
    long long n = (long long)NTOK * DIN;
    cvt_f32_bf16<<<(unsigned)(n / 4 / 256), 256, 0, stream>>>(x, x_bf, n);
  }
  transpose_f32_bf16<<<dim3(DH / 32, DIN / 32, 1),     256, 0, stream>>>(W_in,  WinT,  DIN,  DH);
  transpose_f32_bf16<<<dim3(DMOE / 32, DH / 32, NEXP), 256, 0, stream>>>(W1,    W1T,   DH,   DMOE);
  transpose_f32_bf16<<<dim3(DH / 32, DMOE / 32, NEXP), 256, 0, stream>>>(W2,    W2T,   DMOE, DH);
  transpose_f32_bf16<<<dim3(DOUT / 32, DH / 32, 1),    256, 0, stream>>>(W_out, WoutT, DH,   DOUT);

  // --- GEMM0: h = x @ W_in + b_in -----------------------------------------
  gemm_bf16<0><<<dim3(NTOK / 128, DH / 128, 1), 256, 0, stream>>>(
      x_bf, WinT, b_in, h_bf, NTOK, DH, DIN, 0, 0,
      nullptr, nullptr, nullptr, nullptr, nullptr);

  // --- gating + routing ----------------------------------------------------
  gate_kernel<<<NTOK / 8, 256, 0, stream>>>(h_bf, W_g, b_g, topi, topw, counts);
  scan_kernel<<<1, 32, 0, stream>>>(counts, offsets, counters);
  assign_kernel<<<NTOK / 256, 256, 0, stream>>>(topi, topw, offsets, counters, tokOf, wOf);

  // --- FFN1 (gathered, top-2 only): mid = relu(h W1 + b1) ------------------
  gemm_bf16<1><<<dim3(NTOK / 128, DMOE / 128, NEXP), 256, 0, stream>>>(
      h_bf, W1T, b1, mid_bf, NTOK, DMOE, DH,
      (long long)DMOE * DH, DMOE, counts, offsets, tokOf, wOf, nullptr);

  // --- FFN2: moe[tok] += w * (mid W2 + b2) ---------------------------------
  gemm_bf16<2><<<dim3(NTOK / 128, DH / 128, NEXP), 256, 0, stream>>>(
      mid_bf, W2T, b2, nullptr, NTOK, DH, DMOE,
      (long long)DH * DMOE, DH, counts, offsets, tokOf, wOf, moe);

  // --- GEMM3: out = moe @ W_out + b_out ------------------------------------
  {
    long long n = (long long)NTOK * DH;
    cvt_f32_bf16<<<(unsigned)(n / 4 / 256), 256, 0, stream>>>(moe, moe_bf, n);
  }
  gemm_bf16<3><<<dim3(NTOK / 128, DOUT / 128, 1), 256, 0, stream>>>(
      moe_bf, WoutT, b_out, d_out, NTOK, DOUT, DH, 0, 0,
      nullptr, nullptr, nullptr, nullptr, nullptr);
}
